// NormalCell_4243427689032
// MI455X (gfx1250) — matrix operand order repaired and checked by reading, not compiled, not run on hardware
//
#include <hip/hip_runtime.h>

typedef _Float16 half_t;
typedef __attribute__((ext_vector_type(16))) _Float16 v16h;
typedef __attribute__((ext_vector_type(8)))  _Float16 v8h;
typedef __attribute__((ext_vector_type(8)))  float    v8f;

#define DEV __device__ __forceinline__

// ---------------- problem constants ----------------
constexpr int Bn   = 32;
constexpr int Nn   = 196;
constexpr int Cc   = 768;
constexpr int Hh   = 12;
constexpr int HDd  = 64;
constexpr int HIDd = 3072;
constexpr int Mrows = Bn * Nn;   // 6272
constexpr int BH   = Bn * Hh;    // 384
constexpr int NP   = 208;        // padded seq len (13*16)
constexpr int KP   = 224;        // padded key-K   (7*32)

// ---------------- WMMA fragment loaders ----------------
// A-matrix 16x32 f16 layout (ISA 7.12.2): lanes 0-15 = rows, K groups of 8:
//   lanes 0-15 : K=[0..7] in v0-3, K=[16..23] in v4-7
//   lanes16-31 : K=[8..15],        K=[24..31]
DEV v16h load_a_tile(const half_t* __restrict__ base, int ld,
                     int rowBase, int k0, int lane, int rowClamp) {
  int r = rowBase + (lane & 15);
  r = (r < rowClamp) ? r : (rowClamp - 1);
  const half_t* p = base + (size_t)r * ld + k0 + ((lane >> 4) << 3);
  union { v16h v; v8h h[2]; } u;
  u.h[0] = *(const v8h*)p;
  u.h[1] = *(const v8h*)(p + 16);
  return u.v;
}

// B-matrix 32x16 f16 layout (same K order as A):
//   lanes 0-15 hold K=[0..7],[16..23]; lanes 16-31 hold K=[8..15],[24..31].
// We load rows of B^T (i.e. the [N,K] weight matrix) per lane.
DEV v16h load_b_tile(const half_t* __restrict__ base, int ld,
                     int rowBase, int k0, int lane, int rowClamp) {
  int r = rowBase + (lane & 15);
  r = (r < rowClamp) ? r : (rowClamp - 1);
  const half_t* p = base + (size_t)r * ld + k0 + ((lane >> 4) << 3);
  union { v16h v; v8h h[2]; } u;
  u.h[0] = *(const v8h*)p;
  u.h[1] = *(const v8h*)(p + 16);
  return u.v;
}

DEV v16h load_a_or_zero(const half_t* __restrict__ rowPtr, int k0, int lane, bool valid) {
  union { v16h v; v8h h[2]; } u;
  if (valid) {
    const half_t* p = rowPtr + k0 + ((lane >> 4) << 3);
    u.h[0] = *(const v8h*)p;
    u.h[1] = *(const v8h*)(p + 16);
  } else {
    u.v = v16h{};
  }
  return u.v;
}

// f32 C/D 16x16 layout: lanes0-15 N=lane M=vgpr; lanes16-31 N=lane-16 M=8+vgpr
DEV void store_tile(float* __restrict__ C, int ldc, const v8f& acc,
                    int rowTile, int colTile, int lane, int Mv, int Nv,
                    const float* __restrict__ bias) {
  int nI = colTile + (lane & 15);
  if (nI >= Nv) return;
  int mB = rowTile + ((lane >> 4) << 3);
  float bv = bias ? bias[nI] : 0.0f;
#pragma unroll
  for (int e = 0; e < 8; ++e) {
    int m = mB + e;
    if (m < Mv) C[(size_t)m * ldc + nI] = acc[e] + bv;
  }
}

#define WMMA_F16(A_, B_, C_) \
  __builtin_amdgcn_wmma_f32_16x16x32_f16(false, (A_), false, (B_), (short)0, (C_), false, false)

// ---------------- generic batched GEMM: C = A(MxK) * B(NxK)^T (+bias) ----------------
// 128 threads = 4 waves; each wave computes a 32x64 tile (2x4 WMMA subtiles);
// block tile = 128 (M) x 64 (N).
__global__ __launch_bounds__(128)
void gemm_f16_abT(const half_t* __restrict__ A, const half_t* __restrict__ Bm,
                  float* __restrict__ C, const float* __restrict__ bias,
                  int M, int N, int K, int lda, int ldb, int ldc,
                  int Mclamp, int Nclamp,
                  long long sA, long long sB, long long sC) {
  long long z = blockIdx.z;
  const half_t* Ab = A + (size_t)(z * sA);
  const half_t* Bb = Bm + (size_t)(z * sB);
  float* Cb = C + (size_t)(z * sC);
  int lane = threadIdx.x & 31;
  int wave = threadIdx.x >> 5;            // 0..3 along M
  int rowBase = blockIdx.y * 128 + wave * 32;
  int colBase = blockIdx.x * 64;
  v8f acc[2][4] = {};
  for (int k0 = 0; k0 < K; k0 += 32) {
    v16h a0 = load_a_tile(Ab, lda, rowBase,      k0, lane, Mclamp);
    v16h a1 = load_a_tile(Ab, lda, rowBase + 16, k0, lane, Mclamp);
    v16h b0 = load_b_tile(Bb, ldb, colBase,      k0, lane, Nclamp);
    v16h b1 = load_b_tile(Bb, ldb, colBase + 16, k0, lane, Nclamp);
    v16h b2 = load_b_tile(Bb, ldb, colBase + 32, k0, lane, Nclamp);
    v16h b3 = load_b_tile(Bb, ldb, colBase + 48, k0, lane, Nclamp);
    if (k0 + 32 < K) {  // pull next K-step lines toward the WGP (global_prefetch)
      int ra = rowBase + (lane & 15); ra = (ra < Mclamp) ? ra : Mclamp - 1;
      int rb = colBase + (lane & 15); rb = (rb < Nclamp) ? rb : Nclamp - 1;
      __builtin_prefetch((const void*)(Ab + (size_t)ra * lda + k0 + 32), 0, 3);
      __builtin_prefetch((const void*)(Bb + (size_t)rb * ldb + k0 + 32), 0, 3);
    }
    acc[0][0] = WMMA_F16(a0, b0, acc[0][0]);
    acc[0][1] = WMMA_F16(a0, b1, acc[0][1]);
    acc[0][2] = WMMA_F16(a0, b2, acc[0][2]);
    acc[0][3] = WMMA_F16(a0, b3, acc[0][3]);
    acc[1][0] = WMMA_F16(a1, b0, acc[1][0]);
    acc[1][1] = WMMA_F16(a1, b1, acc[1][1]);
    acc[1][2] = WMMA_F16(a1, b2, acc[1][2]);
    acc[1][3] = WMMA_F16(a1, b3, acc[1][3]);
  }
#pragma unroll
  for (int i = 0; i < 2; ++i)
#pragma unroll
    for (int j = 0; j < 4; ++j)
      store_tile(Cb, ldc, acc[i][j], rowBase + i * 16, colBase + j * 16, lane, M, N, bias);
}

// ---------------- 3x3 SAME conv as implicit GEMM over 9 taps ----------------
// In: channel-last f16 [B,196,Cin]; Wt: tap-major f16 [9][Cout][Cin]
__global__ __launch_bounds__(128)
void conv3x3_gemm(const half_t* __restrict__ In, const half_t* __restrict__ Wt,
                  float* __restrict__ Out, const float* __restrict__ bias,
                  int Cin, int Cout) {
  int lane = threadIdx.x & 31;
  int wave = threadIdx.x >> 5;
  int rowBase = blockIdx.y * 128 + wave * 32;
  int colBase = blockIdx.x * 64;
  v8f acc[2][4] = {};
  int r0 = rowBase + (lane & 15);
  int r1 = r0 + 16;
  int b0i = r0 / 196, p0i = r0 % 196, y0 = p0i / 14, x0 = p0i % 14;
  int b1i = r1 / 196, p1i = r1 % 196, y1 = p1i / 14, x1 = p1i % 14;
  for (int t = 0; t < 9; ++t) {
    int dy = t / 3 - 1, dx = t % 3 - 1;
    const half_t* w = Wt + (size_t)t * Cout * Cin;
    int yy0 = y0 + dy, xx0 = x0 + dx;
    int yy1 = y1 + dy, xx1 = x1 + dx;
    bool v0 = ((unsigned)yy0 < 14u) && ((unsigned)xx0 < 14u);
    bool v1 = ((unsigned)yy1 < 14u) && ((unsigned)xx1 < 14u);
    const half_t* p0 = In + ((size_t)b0i * 196 + yy0 * 14 + xx0) * Cin;
    const half_t* p1 = In + ((size_t)b1i * 196 + yy1 * 14 + xx1) * Cin;
    for (int k0 = 0; k0 < Cin; k0 += 32) {
      v16h a0 = load_a_or_zero(p0, k0, lane, v0);
      v16h a1 = load_a_or_zero(p1, k0, lane, v1);
      v16h b0 = load_b_tile(w, Cin, colBase,      k0, lane, Cout);
      v16h b1 = load_b_tile(w, Cin, colBase + 16, k0, lane, Cout);
      v16h b2 = load_b_tile(w, Cin, colBase + 32, k0, lane, Cout);
      v16h b3 = load_b_tile(w, Cin, colBase + 48, k0, lane, Cout);
      acc[0][0] = WMMA_F16(a0, b0, acc[0][0]);
      acc[0][1] = WMMA_F16(a0, b1, acc[0][1]);
      acc[0][2] = WMMA_F16(a0, b2, acc[0][2]);
      acc[0][3] = WMMA_F16(a0, b3, acc[0][3]);
      acc[1][0] = WMMA_F16(a1, b0, acc[1][0]);
      acc[1][1] = WMMA_F16(a1, b1, acc[1][1]);
      acc[1][2] = WMMA_F16(a1, b2, acc[1][2]);
      acc[1][3] = WMMA_F16(a1, b3, acc[1][3]);
    }
  }
#pragma unroll
  for (int i = 0; i < 2; ++i)
#pragma unroll
    for (int j = 0; j < 4; ++j)
      store_tile(Out, Cout, acc[i][j], rowBase + i * 16, colBase + j * 16, lane,
                 Mrows, Cout, bias);
}

// ---------------- LayerNorm (C=768) -> f16 ----------------
__global__ __launch_bounds__(256)
void ln_f16_kernel(const float* __restrict__ X, const float* __restrict__ g,
                   const float* __restrict__ b, half_t* __restrict__ Y) {
  int row = blockIdx.x;
  int t = threadIdx.x;
  const float* xr = X + (size_t)row * Cc;
  float x0 = xr[t], x1 = xr[t + 256], x2 = xr[t + 512];
  __shared__ float rs[256], rq[256];
  rs[t] = x0 + x1 + x2;
  rq[t] = x0 * x0 + x1 * x1 + x2 * x2;
  __syncthreads();
  for (int o = 128; o > 0; o >>= 1) {
    if (t < o) { rs[t] += rs[t + o]; rq[t] += rq[t + o]; }
    __syncthreads();
  }
  float mean = rs[0] / Cc;
  float var  = rq[0] / Cc - mean * mean;
  float inv  = rsqrtf(var + 1e-5f);
  half_t* yr = Y + (size_t)row * Cc;
  yr[t]       = (half_t)((x0 - mean) * inv * g[t]       + b[t]);
  yr[t + 256] = (half_t)((x1 - mean) * inv * g[t + 256] + b[t + 256]);
  yr[t + 512] = (half_t)((x2 - mean) * inv * g[t + 512] + b[t + 512]);
}

// ---------------- elementwise kernels ----------------
__global__ void cvt_f16(const float* __restrict__ in, half_t* __restrict__ out, size_t n) {
  size_t i = (size_t)blockIdx.x * blockDim.x + threadIdx.x;
  if (i < n) out[i] = (half_t)in[i];
}

// conv weight [O][I][3][3] f32 -> tap-major [9][O][I] f16
__global__ void permute_conv_w(const float* __restrict__ w, half_t* __restrict__ o,
                               int Cout, int Cin) {
  size_t i = (size_t)blockIdx.x * blockDim.x + threadIdx.x;
  size_t total = (size_t)9 * Cout * Cin;
  if (i >= total) return;
  int ci = (int)(i % Cin);
  size_t t2 = i / Cin;
  int co = (int)(t2 % Cout);
  int tap = (int)(t2 / Cout);
  o[i] = (half_t)w[((size_t)co * Cin + ci) * 9 + tap];
}

// qkv [6272, 2304] f32 -> q(scaled)/k padded [bh][208][64] f16, v^T [bh][64][224] f16
__global__ void qkv_split(const float* __restrict__ qkv, half_t* __restrict__ qh,
                          half_t* __restrict__ kh, half_t* __restrict__ vt) {
  size_t idx = (size_t)blockIdx.x * blockDim.x + threadIdx.x;
  size_t total = (size_t)Mrows * 3 * Cc;
  if (idx >= total) return;
  int col = (int)(idx % (3 * Cc));
  size_t row = idx / (3 * Cc);
  int b = (int)(row / Nn), n = (int)(row % Nn);
  int s = col / Cc;
  int rem = col % Cc;
  int h = rem / HDd, d = rem % HDd;
  int bh = b * Hh + h;
  float v = qkv[idx];
  if (s == 0)       qh[((size_t)bh * NP + n) * HDd + d] = (half_t)(v * 0.125f);
  else if (s == 1)  kh[((size_t)bh * NP + n) * HDd + d] = (half_t)v;
  else              vt[((size_t)bh * HDd + d) * KP + n] = (half_t)v;
}

// rel_h[bh][q][kh] = sum_d q*Rh[qh-kh+13,d] ; rel_w analog
__global__ void rel_bias_kernel(const half_t* __restrict__ qh,
                                const float* __restrict__ tabH,
                                const float* __restrict__ tabW,
                                float* __restrict__ relH, float* __restrict__ relW) {
  size_t idx = (size_t)blockIdx.x * blockDim.x + threadIdx.x;
  size_t total = (size_t)BH * Nn * 14;
  if (idx >= total) return;
  int j = (int)(idx % 14);
  size_t t2 = idx / 14;
  int q = (int)(t2 % Nn);
  int bh = (int)(t2 / Nn);
  int qy = q / 14, qx = q % 14;
  const half_t* qp = qh + ((size_t)bh * NP + q) * HDd;
  const float* th = tabH + (size_t)(qy - j + 13) * HDd;
  const float* tw = tabW + (size_t)(qx - j + 13) * HDd;
  float sh = 0.f, sw = 0.f;
#pragma unroll 8
  for (int d = 0; d < HDd; ++d) {
    float qv = (float)qp[d];
    sh += qv * th[d];
    sw += qv * tw[d];
  }
  relH[idx] = sh;
  relW[idx] = sw;
}

// softmax over 196 keys with rel bias; scores [bh][208][208] f32 -> attn [bh][208][224] f16
__global__ __launch_bounds__(256)
void softmax_bias(const float* __restrict__ scores, const float* __restrict__ relH,
                  const float* __restrict__ relW, half_t* __restrict__ attn) {
  int row = blockIdx.x;               // 0 .. BH*196-1
  int bh = row / Nn, q = row % Nn;
  const float* s  = scores + ((size_t)bh * NP + q) * NP;
  const float* rh = relH + ((size_t)bh * Nn + q) * 14;
  const float* rw = relW + ((size_t)bh * Nn + q) * 14;
  half_t* a = attn + ((size_t)bh * NP + q) * KP;
  int t = threadIdx.x;
  bool active = t < Nn;
  float val = 0.f, v = -3.0e38f;
  if (active) {
    int kh = t / 14, kw = t % 14;
    val = s[t] + rh[kh] + rw[kw];
    v = val;
  }
  __shared__ float sm[256];
  sm[t] = v; __syncthreads();
  for (int o = 128; o > 0; o >>= 1) { if (t < o) sm[t] = fmaxf(sm[t], sm[t + o]); __syncthreads(); }
  float mx = sm[0]; __syncthreads();
  float e = active ? expf(val - mx) : 0.f;
  sm[t] = e; __syncthreads();
  for (int o = 128; o > 0; o >>= 1) { if (t < o) sm[t] += sm[t + o]; __syncthreads(); }
  float inv = 1.f / sm[0];
  if (active) a[t] = (half_t)(e * inv);
}

// attnV [bh][208][64] f32 -> xa [b*196, 768] f16
__global__ void merge_heads(const float* __restrict__ av, half_t* __restrict__ xa) {
  size_t idx = (size_t)blockIdx.x * blockDim.x + threadIdx.x;
  size_t total = (size_t)Mrows * Cc;
  if (idx >= total) return;
  int col = (int)(idx % Cc);
  size_t row = idx / Cc;
  int b = (int)(row / Nn), n = (int)(row % Nn);
  int h = col / HDd, d = col % HDd;
  xa[idx] = (half_t)av[(((size_t)(b * Hh + h)) * NP + n) * HDd + d];
}

// eval-mode BN + SiLU -> f16 (channel-last [6272, 3072])
__global__ void bn_silu(const float* __restrict__ h1, const float* __restrict__ g,
                        const float* __restrict__ b, const float* __restrict__ mean,
                        const float* __restrict__ var, half_t* __restrict__ o) {
  size_t idx = (size_t)blockIdx.x * blockDim.x + threadIdx.x;
  size_t total = (size_t)Mrows * HIDd;
  if (idx >= total) return;
  int c = (int)(idx % HIDd);
  float t = (h1[idx] - mean[c]) * rsqrtf(var[c] + 1e-5f) * g[c] + b[c];
  o[idx] = (half_t)(t / (1.f + expf(-t)));   // t * sigmoid(t)
}

__global__ void gelu_cvt(const float* __restrict__ in, half_t* __restrict__ out, size_t n) {
  size_t i = (size_t)blockIdx.x * blockDim.x + threadIdx.x;
  if (i < n) {
    float x = in[i];
    out[i] = (half_t)(0.5f * x * (1.f + erff(x * 0.70710678118654752f)));
  }
}

__global__ void add3(const float* __restrict__ a, const float* __restrict__ b,
                     const float* __restrict__ c, float* __restrict__ o, size_t n) {
  size_t i = (size_t)blockIdx.x * blockDim.x + threadIdx.x;
  if (i < n) o[i] = a[i] + b[i] + c[i];
}

__global__ void add2(const float* __restrict__ a, const float* __restrict__ b,
                     float* __restrict__ o, size_t n) {
  size_t i = (size_t)blockIdx.x * blockDim.x + threadIdx.x;
  if (i < n) o[i] = a[i] + b[i];
}

// ---------------- host side ----------------
static void launch_gemm(const half_t* A, const half_t* B, float* C, const float* bias,
                        int M, int N, int K, int lda, int ldb, int ldc,
                        int Mclamp, int Nclamp,
                        long long sA, long long sB, long long sC,
                        int batch, hipStream_t st) {
  dim3 g((unsigned)((N + 63) / 64), (unsigned)((M + 127) / 128), (unsigned)batch);
  gemm_f16_abT<<<g, 128, 0, st>>>(A, B, C, bias, M, N, K, lda, ldb, ldc,
                                  Mclamp, Nclamp, sA, sB, sC);
}

extern "C" void kernel_launch(void* const* d_in, const int* in_sizes, int n_in,
                              void* d_out, int out_size, void* d_ws, size_t ws_size,
                              hipStream_t stream) {
  (void)in_sizes; (void)n_in; (void)out_size; (void)ws_size;
  const float* x      = (const float*)d_in[0];
  const float* n1g    = (const float*)d_in[1];
  const float* n1b    = (const float*)d_in[2];
  const float* qkv_w  = (const float*)d_in[3];
  const float* rph    = (const float*)d_in[4];
  const float* rpw    = (const float*)d_in[5];
  const float* proj_w = (const float*)d_in[6];
  const float* proj_b = (const float*)d_in[7];
  const float* n2g    = (const float*)d_in[8];
  const float* n2b    = (const float*)d_in[9];
  const float* fc1_w  = (const float*)d_in[10];
  const float* fc1_b  = (const float*)d_in[11];
  const float* fc2_w  = (const float*)d_in[12];
  const float* fc2_b  = (const float*)d_in[13];
  const float* c1w    = (const float*)d_in[14];
  const float* c1b    = (const float*)d_in[15];
  const float* bng    = (const float*)d_in[16];
  const float* bnb    = (const float*)d_in[17];
  const float* bnm    = (const float*)d_in[18];
  const float* bnv    = (const float*)d_in[19];
  const float* c2w    = (const float*)d_in[20];
  const float* c2b    = (const float*)d_in[21];
  float* out = (float*)d_out;

  char* ws = (char*)d_ws;
  size_t off = 0;
  auto carve = [&](size_t bytes) -> void* {
    void* p = ws + off;
    off += (bytes + 255) & ~(size_t)255;
    return p;
  };

  half_t* x_h     = (half_t*)carve((size_t)Mrows * Cc * 2);
  half_t* xn_h    = (half_t*)carve((size_t)Mrows * Cc * 2);
  half_t* qkvw_h  = (half_t*)carve((size_t)3 * Cc * Cc * 2);
  half_t* projw_h = (half_t*)carve((size_t)Cc * Cc * 2);
  half_t* fc1w_h  = (half_t*)carve((size_t)HIDd * Cc * 2);
  half_t* fc2w_h  = (half_t*)carve((size_t)Cc * HIDd * 2);
  half_t* c1w_h   = (half_t*)carve((size_t)9 * HIDd * Cc * 2);
  half_t* c2w_h   = (half_t*)carve((size_t)9 * Cc * HIDd * 2);
  half_t* q_h     = (half_t*)carve((size_t)BH * NP * HDd * 2);
  half_t* k_h     = (half_t*)carve((size_t)BH * NP * HDd * 2);
  half_t* v_t     = (half_t*)carve((size_t)BH * HDd * KP * 2);
  float*  relH    = (float*) carve((size_t)BH * Nn * 14 * 4);
  float*  relW    = (float*) carve((size_t)BH * Nn * 14 * 4);
  float*  attnV   = (float*) carve((size_t)BH * NP * HDd * 4);
  half_t* xa_h    = (half_t*)carve((size_t)Mrows * Cc * 2);
  float*  projO   = (float*) carve((size_t)Mrows * Cc * 4);
  float*  convO   = (float*) carve((size_t)Mrows * Cc * 4);
  float*  xsum    = (float*) carve((size_t)Mrows * Cc * 4);
  half_t* xm_h    = (half_t*)carve((size_t)Mrows * Cc * 2);
  float*  fc2O    = (float*) carve((size_t)Mrows * Cc * 4);
  void*   arenaA  = carve((size_t)Mrows * HIDd * 4);  // qkvO / scores / h1f32 / fc1O
  void*   arenaB  = carve((size_t)Mrows * HIDd * 2);  // attn / h1h / fc1h
  float*  qkvO   = (float*)arenaA;
  float*  scores = (float*)arenaA;
  float*  h1f    = (float*)arenaA;
  float*  fc1O   = (float*)arenaA;
  half_t* attnP  = (half_t*)arenaB;
  half_t* h1h    = (half_t*)arenaB;
  half_t* fc1h   = (half_t*)arenaB;

  auto ew = [](size_t n) { return dim3((unsigned)((n + 255) / 256)); };

  // --- weight / input precision conversion ---
  cvt_f16<<<ew((size_t)3 * Cc * Cc), 256, 0, stream>>>(qkv_w, qkvw_h, (size_t)3 * Cc * Cc);
  cvt_f16<<<ew((size_t)Cc * Cc), 256, 0, stream>>>(proj_w, projw_h, (size_t)Cc * Cc);
  cvt_f16<<<ew((size_t)HIDd * Cc), 256, 0, stream>>>(fc1_w, fc1w_h, (size_t)HIDd * Cc);
  cvt_f16<<<ew((size_t)Cc * HIDd), 256, 0, stream>>>(fc2_w, fc2w_h, (size_t)Cc * HIDd);
  permute_conv_w<<<ew((size_t)9 * HIDd * Cc), 256, 0, stream>>>(c1w, c1w_h, HIDd, Cc);
  permute_conv_w<<<ew((size_t)9 * Cc * HIDd), 256, 0, stream>>>(c2w, c2w_h, Cc, HIDd);
  cvt_f16<<<ew((size_t)Mrows * Cc), 256, 0, stream>>>(x, x_h, (size_t)Mrows * Cc);

  // --- zero padded attention buffers ---
  hipMemsetAsync(q_h,   0, (size_t)BH * NP * HDd * 2, stream);
  hipMemsetAsync(k_h,   0, (size_t)BH * NP * HDd * 2, stream);
  hipMemsetAsync(v_t,   0, (size_t)BH * HDd * KP * 2, stream);
  hipMemsetAsync(attnP, 0, (size_t)BH * NP * KP * 2, stream);

  // --- attention branch ---
  ln_f16_kernel<<<Mrows, 256, 0, stream>>>(x, n1g, n1b, xn_h);
  launch_gemm(xn_h, qkvw_h, qkvO, nullptr, Mrows, 3 * Cc, Cc, Cc, Cc, 3 * Cc,
              Mrows, 3 * Cc, 0, 0, 0, 1, stream);
  qkv_split<<<ew((size_t)Mrows * 3 * Cc), 256, 0, stream>>>(qkvO, q_h, k_h, v_t);
  rel_bias_kernel<<<ew((size_t)BH * Nn * 14), 256, 0, stream>>>(q_h, rph, rpw, relH, relW);
  launch_gemm(q_h, k_h, scores, nullptr, NP, NP, HDd, HDd, HDd, NP,
              NP, NP, (long long)NP * HDd, (long long)NP * HDd, (long long)NP * NP,
              BH, stream);
  softmax_bias<<<BH * Nn, 256, 0, stream>>>(scores, relH, relW, attnP);
  launch_gemm(attnP, v_t, attnV, nullptr, NP, HDd, KP, KP, KP, HDd,
              NP, HDd, (long long)NP * KP, (long long)HDd * KP, (long long)NP * HDd,
              BH, stream);
  merge_heads<<<ew((size_t)Mrows * Cc), 256, 0, stream>>>(attnV, xa_h);
  launch_gemm(xa_h, projw_h, projO, proj_b, Mrows, Cc, Cc, Cc, Cc, Cc,
              Mrows, Cc, 0, 0, 0, 1, stream);

  // --- conv (PCM) branch on pre-attention x ---
  dim3 cg1((unsigned)(HIDd / 64), (unsigned)(Mrows / 128));
  conv3x3_gemm<<<cg1, 128, 0, stream>>>(x_h, c1w_h, h1f, c1b, Cc, HIDd);
  bn_silu<<<ew((size_t)Mrows * HIDd), 256, 0, stream>>>(h1f, bng, bnb, bnm, bnv, h1h);
  dim3 cg2((unsigned)(Cc / 64), (unsigned)(Mrows / 128));
  conv3x3_gemm<<<cg2, 128, 0, stream>>>(h1h, c2w_h, convO, c2b, HIDd, Cc);

  // --- residuals + MLP ---
  add3<<<ew((size_t)Mrows * Cc), 256, 0, stream>>>(x, projO, convO, xsum, (size_t)Mrows * Cc);
  ln_f16_kernel<<<Mrows, 256, 0, stream>>>(xsum, n2g, n2b, xm_h);
  launch_gemm(xm_h, fc1w_h, fc1O, fc1_b, Mrows, HIDd, Cc, Cc, Cc, HIDd,
              Mrows, HIDd, 0, 0, 0, 1, stream);
  gelu_cvt<<<ew((size_t)Mrows * HIDd), 256, 0, stream>>>(fc1O, fc1h, (size_t)Mrows * HIDd);
  launch_gemm(fc1h, fc2w_h, fc2O, fc2_b, Mrows, Cc, HIDd, HIDd, HIDd, Cc,
              Mrows, Cc, 0, 0, 0, 1, stream);
  add2<<<ew((size_t)Mrows * Cc), 256, 0, stream>>>(xsum, fc2O, out, (size_t)Mrows * Cc);
}
